// GPT2SelfAttention_76235669504238
// MI455X (gfx1250) — matrix operand order repaired and checked by reading, not compiled, not run on hardware
//
#include <hip/hip_runtime.h>

// ---------------------------------------------------------------------------
// GPT-2 self-attention for MI455X (gfx1250), wave32 + WMMA bf16.
//   B=4, T=2048, C=1024, H=16, hd=64
// Pipeline: qkv_gemm -> flash_attn -> proj_gemm, intermediates bf16 in d_ws.
// CDNA5 paths: v_wmma_f32_16x16x32_bf16 cores, TDM tensor_load_to_lds for the
// Q tile (LDS row padding via D# pad fields), global_load_async_to_lds_b128
// for the K-tile stream, global_prefetch_b8 for GEMM weight streaming.
// ---------------------------------------------------------------------------

#define BB   4
#define TT   2048
#define CC   1024
#define NH   16
#define HD   64
#define C3   3072
#define MTOT (BB * TT)          // 8192 rows

typedef __bf16 bf16_t;
typedef __attribute__((ext_vector_type(16))) __bf16 v16bf;
typedef __attribute__((ext_vector_type(8)))  __bf16 v8bf;
typedef __attribute__((ext_vector_type(8)))  float  v8f;
typedef __attribute__((ext_vector_type(4)))  float  v4f;
typedef __attribute__((ext_vector_type(4)))  unsigned u32x4;
typedef __attribute__((ext_vector_type(8)))  int      i32x8;
typedef __attribute__((ext_vector_type(4)))  int      i32x4;

__device__ inline bf16_t f2bf(float f) {
  unsigned u = __builtin_bit_cast(unsigned, f);
  unsigned r = u + 0x7FFFu + ((u >> 16) & 1u);          // round-to-nearest-even
  unsigned short h = (unsigned short)(r >> 16);
  return __builtin_bit_cast(bf16_t, h);
}

__device__ inline v8f vzero8() {
  v8f z;
#pragma unroll
  for (int i = 0; i < 8; ++i) z[i] = 0.0f;
  return z;
}

__device__ inline v8f wmma_bf16(v16bf a, v16bf b, v8f c) {
  return __builtin_amdgcn_wmma_f32_16x16x32_bf16(false, a, false, b, (short)0, c,
                                                 false, false);
}

// A-matrix fragment (16x32 bf16), ISA layout:
//   elements 0..7  = K[h*8 .. h*8+7], elements 8..15 = K[16+h*8 .. +7]
__device__ inline v16bf load_frag_a(const bf16_t* rowPtr, int half) {
  v8bf lo = *(const v8bf*)(rowPtr + half * 8);
  v8bf hi = *(const v8bf*)(rowPtr + 16 + half * 8);
  v16bf r;
#pragma unroll
  for (int i = 0; i < 8; ++i) { r[i] = lo[i]; r[i + 8] = hi[i]; }
  return r;
}

// B-matrix fragment (32x16 bf16): elements 0..7 = K[h*8 .. h*8+7], elements 8..15 = K[16+h*8 .. +7] for column n.
__device__ inline v16bf load_frag_b(const bf16_t* colPtr, int half) {
  v8bf lo = *(const v8bf*)(colPtr + half * 8);
  v8bf hi = *(const v8bf*)(colPtr + 16 + half * 8);
  v16bf r;
#pragma unroll
  for (int i = 0; i < 8; ++i) { r[i] = lo[i]; r[i + 8] = hi[i]; }
  return r;
}

// Async memory->LDS copy, 16B per lane (ASYNCcnt). CDNA5 VGLOBAL async op.
__device__ inline void async_copy_b128(bf16_t* ldsDst, const bf16_t* gsrc) {
  unsigned ldsOff = (unsigned)(size_t)ldsDst;   // LDS aperture low bits
  asm volatile("global_load_async_to_lds_b128 %0, %1, off"
               :: "v"(ldsOff), "v"(gsrc) : "memory");
}
__device__ inline void wait_asynccnt0() {
  asm volatile("s_wait_asynccnt 0" ::: "memory");
}

// ---------------------------------------------------------------------------
// Kernel 1: qkv = x @ w_attn + b_attn ; scatter into Q (pre-scaled), K, V
// ---------------------------------------------------------------------------
#define G_BM 128
#define G_BK 32
#define G_LD (G_BK + 8)   // 40 bf16, 80 B rows (16B aligned)

__global__ __launch_bounds__(256) void qkv_gemm(
    const float* __restrict__ x, const float* __restrict__ w,
    const float* __restrict__ bias,
    bf16_t* __restrict__ Qd, bf16_t* __restrict__ Kd, bf16_t* __restrict__ Vd) {
  __shared__ bf16_t sA[G_BM * G_LD];        // [m][k]
  __shared__ bf16_t sB[G_BM * G_LD];        // [n][k]  (transposed weights)

  const int tid  = threadIdx.x;
  const int lane = tid & 31, wave = tid >> 5;
  const int wm = wave & 3, wn = wave >> 2;   // 4 waves along M, 2 along N
  const int half = lane >> 4, nn = lane & 15;
  const int m0 = blockIdx.x * G_BM;
  const int n0 = blockIdx.y * G_BM;

  v8f acc[2][4];
#pragma unroll
  for (int i = 0; i < 2; ++i)
#pragma unroll
    for (int j = 0; j < 4; ++j) acc[i][j] = vzero8();

  for (int k0 = 0; k0 < CC; k0 += G_BK) {
    // warm L2 for the next K-slice (global_prefetch_b8)
    if (k0 + G_BK < CC) {
      __builtin_prefetch(x + (size_t)(m0 + (tid >> 1)) * CC + k0 + G_BK, 0, 0);
      __builtin_prefetch(w + (size_t)(k0 + G_BK + (tid >> 3)) * C3 + n0 + (tid & 7) * 16, 0, 0);
    }
    __syncthreads();
    { // stage A: 128x32 f32 -> bf16
      const int r = tid >> 1, c = (tid & 1) * 16;
      const float* src = x + (size_t)(m0 + r) * CC + k0 + c;
      bf16_t* dst = sA + r * G_LD + c;
#pragma unroll
      for (int i = 0; i < 16; i += 4) {
        v4f f = *(const v4f*)(src + i);
#pragma unroll
        for (int q = 0; q < 4; ++q) dst[i + q] = f2bf(f[q]);
      }
    }
    { // stage B transposed: w[k0+kk][n0+c .. +15] -> sB[n][k]
      const int kk = tid >> 3, c = (tid & 7) * 16;
      const float* src = w + (size_t)(k0 + kk) * C3 + n0 + c;
#pragma unroll
      for (int i = 0; i < 16; i += 4) {
        v4f f = *(const v4f*)(src + i);
#pragma unroll
        for (int q = 0; q < 4; ++q) sB[(c + i + q) * G_LD + kk] = f2bf(f[q]);
      }
    }
    __syncthreads();

    v16bf af[2];
#pragma unroll
    for (int i = 0; i < 2; ++i)
      af[i] = load_frag_a(sA + (wm * 32 + i * 16 + nn) * G_LD, half);
#pragma unroll
    for (int j = 0; j < 4; ++j) {
      v16bf bf_ = load_frag_b(sB + (wn * 64 + j * 16 + nn) * G_LD, half);
#pragma unroll
      for (int i = 0; i < 2; ++i) acc[i][j] = wmma_bf16(af[i], bf_, acc[i][j]);
    }
  }

  const float qscale = 0.125f;  // 1/sqrt(64)
#pragma unroll
  for (int i = 0; i < 2; ++i)
#pragma unroll
    for (int j = 0; j < 4; ++j)
#pragma unroll
      for (int r = 0; r < 8; ++r) {
        const int mg = m0 + wm * 32 + i * 16 + half * 8 + r;
        const int ng = n0 + wn * 64 + j * 16 + nn;
        const float v = acc[i][j][r] + bias[ng];
        const int b = mg >> 11, t = mg & 2047;
        const int c = ng & (CC - 1);
        const int h = c >> 6, d = c & 63;
        const size_t off = (((size_t)b * NH + h) * TT + t) * HD + d;
        if (ng < CC)          Qd[off] = f2bf(v * qscale);
        else if (ng < 2 * CC) Kd[off] = f2bf(v);
        else                  Vd[off] = f2bf(v);
      }
}

// ---------------------------------------------------------------------------
// Kernel 2: causal flash attention. Block = (q-tile of 64, b*h). 4 waves,
//           16 query rows each. Online softmax, all matmuls via WMMA.
// ---------------------------------------------------------------------------
#define F_LD (HD + 8)   // 72 bf16 rows (144 B, 16B aligned)

__global__ __launch_bounds__(128) void flash_attn(
    const bf16_t* __restrict__ Qd, const bf16_t* __restrict__ Kd,
    const bf16_t* __restrict__ Vd, bf16_t* __restrict__ Yd) {
  __shared__ bf16_t sQ[64 * F_LD];
  __shared__ bf16_t sK[64 * F_LD];
  __shared__ bf16_t sVT[HD * F_LD];      // [d][key]
  __shared__ bf16_t sP[4][16 * F_LD];    // per-wave P tile

  const int tid  = threadIdx.x;
  const int lane = tid & 31, wq = tid >> 5;
  const int half = lane >> 4, nn = lane & 15;
  const int qt = blockIdx.x;             // 0..31
  const int bh = blockIdx.y;             // 0..63 (= b*16 + h)
  const size_t base = (size_t)bh * TT * HD;

  // ---- stage Q tile (64x64 bf16) via the Tensor Data Mover ----------------
  // D#: 2D tile 64x64 of 2B elems from a [T][64] tensor; LDS pad_enable pads
  // every 32 DWORDs (one 64-elem row) by 4 DWORDs (8 bf16) -> F_LD=72 stride.
#if __has_builtin(__builtin_amdgcn_tensor_load_to_lds)
  if (wq == 0) {
    const bf16_t* gsrc = Qd + base + (size_t)qt * 64 * HD;
    const unsigned long long ga = (unsigned long long)(size_t)gsrc;
    u32x4 g0;
    g0[0] = 1u;                                   // count=1, user descriptor
    g0[1] = (unsigned)(size_t)sQ;                 // lds_addr
    g0[2] = (unsigned)ga;                         // global_addr[31:0]
    g0[3] = (unsigned)(ga >> 32) | (2u << 30);    // global_addr[56:32] | type=2
    i32x8 g1;
    g1[0] = (1 << 16)      // data_size = 2B
          | (1 << 20)      // pad_enable
          | (4 << 22)      // pad_interval: 32 DWORDs
          | (3 << 25);     // pad_amount: 4 DWORDs
    g1[1] = (HD << 16);    // tensor_dim0 = 64  (bits 79:48)
    g1[2] = (TT << 16);    // tensor_dim1 = 2048 (bits 111:80, low half)
    g1[3] = (64 << 16);    // tile_dim0 = 64    (bits 127:112)
    g1[4] = 64;            // tile_dim1 = 64, tile_dim2 = 0
    g1[5] = HD;            // tensor_dim0_stride = 64
    g1[6] = 0;
    g1[7] = 0;
    i32x4 gz; gz[0] = 0; gz[1] = 0; gz[2] = 0; gz[3] = 0;
#if __clang_major__ >= 23
    i32x8 gz8;
#pragma unroll
    for (int i = 0; i < 8; ++i) gz8[i] = 0;
    __builtin_amdgcn_tensor_load_to_lds(g0, g1, gz, gz, gz8, 0);
#else
    __builtin_amdgcn_tensor_load_to_lds(g0, g1, gz, gz, 0);
#endif
    __builtin_amdgcn_s_wait_tensorcnt(0);
  }
#else
  { // fallback: plain vector copy
    const bf16_t* src = Qd + base + (size_t)qt * 64 * HD;
#pragma unroll
    for (int it = 0; it < 4; ++it) {
      const int idx = tid + it * 128;          // 512 x v8bf
      const int r = idx >> 3, c = (idx & 7) * 8;
      *(v8bf*)(sQ + r * F_LD + c) = *(const v8bf*)(src + r * HD + c);
    }
  }
#endif
  __syncthreads();

  v16bf aq[2];
#pragma unroll
  for (int kk = 0; kk < 2; ++kk)
    aq[kk] = load_frag_a(sQ + (wq * 16 + nn) * F_LD + kk * 32, half);

  float m_i[8], l_i[8];
  v8f o[4];
#pragma unroll
  for (int r = 0; r < 8; ++r) { m_i[r] = -3.0e38f; l_i[r] = 0.0f; }
#pragma unroll
  for (int j = 0; j < 4; ++j) o[j] = vzero8();

  for (int kt = 0; kt <= qt; ++kt) {
    __syncthreads();
    { // stage K tile via async global->LDS copies (ASYNCcnt)
      const bf16_t* src = Kd + base + (size_t)kt * 64 * HD;
#pragma unroll
      for (int it = 0; it < 4; ++it) {
        const int idx = tid + it * 128;
        const int r = idx >> 3, c = (idx & 7) * 8;
        async_copy_b128(sK + r * F_LD + c, src + r * HD + c);
      }
    }
    { // stage V transposed: sVT[d][key]
      const bf16_t* src = Vd + base + (size_t)kt * 64 * HD;
      const int key = tid >> 1, d0 = (tid & 1) * 32;
      const bf16_t* vs = src + key * HD + d0;
#pragma unroll
      for (int i0 = 0; i0 < 32; i0 += 8) {
        v8bf vv = *(const v8bf*)(vs + i0);
#pragma unroll
        for (int q = 0; q < 8; ++q) sVT[(d0 + i0 + q) * F_LD + key] = vv[q];
      }
    }
    wait_asynccnt0();
    __syncthreads();

    // --- S = Q @ K^T (16x64 per wave) ---
    v8f s[4];
#pragma unroll
    for (int j = 0; j < 4; ++j) s[j] = vzero8();
#pragma unroll
    for (int j = 0; j < 4; ++j)
#pragma unroll
      for (int kk = 0; kk < 2; ++kk) {
        v16bf bk = load_frag_b(sK + (j * 16 + nn) * F_LD + kk * 32, half);
        s[j] = wmma_bf16(aq[kk], bk, s[j]);
      }

    // --- causal mask + online softmax (row stats replicated per 16-lane group)
    float p[4][8];
#pragma unroll
    for (int r = 0; r < 8; ++r) {
      const int rowg = qt * 64 + wq * 16 + half * 8 + r;
      float mx = -3.0e38f;
#pragma unroll
      for (int j = 0; j < 4; ++j) {
        const int colg = kt * 64 + j * 16 + nn;
        float v = (colg <= rowg) ? s[j][r] : -3.0e38f;
        s[j][r] = v;
        mx = fmaxf(mx, v);
      }
#pragma unroll
      for (int off = 1; off < 16; off <<= 1)
        mx = fmaxf(mx, __shfl_xor(mx, off, 32));
      const float newm  = fmaxf(m_i[r], mx);
      const float scale = __expf(m_i[r] - newm);
      float sum = 0.0f;
#pragma unroll
      for (int j = 0; j < 4; ++j) {
        const float e = __expf(s[j][r] - newm);
        p[j][r] = e;
        sum += e;
      }
#pragma unroll
      for (int off = 1; off < 16; off <<= 1)
        sum += __shfl_xor(sum, off, 32);
      l_i[r] = l_i[r] * scale + sum;
      m_i[r] = newm;
#pragma unroll
      for (int j = 0; j < 4; ++j) o[j][r] *= scale;
    }

    // --- P (C/D layout) -> per-wave LDS, re-read in A layout ---
#pragma unroll
    for (int j = 0; j < 4; ++j)
#pragma unroll
      for (int r = 0; r < 8; ++r)
        sP[wq][(half * 8 + r) * F_LD + j * 16 + nn] = f2bf(p[j][r]);
    // same-wave DS RAW: enforce LDS drain before fragment reads (CDNA5 DScnt)
    asm volatile("s_wait_dscnt 0" ::: "memory");

    // --- O += P @ V ---
#pragma unroll
    for (int kk = 0; kk < 2; ++kk) {
      v16bf ap = load_frag_a(sP[wq] + nn * F_LD + kk * 32, half);
#pragma unroll
      for (int j = 0; j < 4; ++j) {
        v16bf bv = load_frag_b(sVT + (j * 16 + nn) * F_LD + kk * 32, half);
        o[j] = wmma_bf16(ap, bv, o[j]);
      }
    }
  }

  // --- finalize: y[b, t, h*64+d] = O / l ---
  const int hh = bh & 15, bb = bh >> 4;
#pragma unroll
  for (int j = 0; j < 4; ++j)
#pragma unroll
    for (int r = 0; r < 8; ++r) {
      const int t = qt * 64 + wq * 16 + half * 8 + r;
      const int d = j * 16 + nn;
      const float v = o[j][r] / l_i[r];
      Yd[((size_t)(bb * TT + t) * CC) + hh * HD + d] = f2bf(v);
    }
}

// ---------------------------------------------------------------------------
// Kernel 3: out = y @ w_proj + b_proj   (8192x1024x1024, f32 out)
// ---------------------------------------------------------------------------
__global__ __launch_bounds__(256) void proj_gemm(
    const bf16_t* __restrict__ y, const float* __restrict__ w,
    const float* __restrict__ bias, float* __restrict__ out) {
  __shared__ bf16_t sA[G_BM * G_LD];   // [m][k]
  __shared__ bf16_t sB[G_BM * G_LD];   // [n][k]

  const int tid  = threadIdx.x;
  const int lane = tid & 31, wave = tid >> 5;
  const int wm = wave & 3, wn = wave >> 2;
  const int half = lane >> 4, nn = lane & 15;
  const int m0 = blockIdx.x * G_BM;
  const int n0 = blockIdx.y * G_BM;

  v8f acc[2][4];
#pragma unroll
  for (int i = 0; i < 2; ++i)
#pragma unroll
    for (int j = 0; j < 4; ++j) acc[i][j] = vzero8();

  for (int k0 = 0; k0 < CC; k0 += G_BK) {
    if (k0 + G_BK < CC) {
      __builtin_prefetch(y + (size_t)(m0 + (tid >> 1)) * CC + k0 + G_BK, 0, 0);
      __builtin_prefetch(w + (size_t)(k0 + G_BK + (tid >> 3)) * CC + n0 + (tid & 7) * 16, 0, 0);
    }
    __syncthreads();
    { // stage A (bf16 copy)
      const int r = tid >> 1, c = (tid & 1) * 16;
      const bf16_t* src = y + (size_t)(m0 + r) * CC + k0 + c;
      bf16_t* dst = sA + r * G_LD + c;
      *(v8bf*)(dst)     = *(const v8bf*)(src);
      *(v8bf*)(dst + 8) = *(const v8bf*)(src + 8);
    }
    { // stage B transposed, f32 -> bf16
      const int kk = tid >> 3, c = (tid & 7) * 16;
      const float* src = w + (size_t)(k0 + kk) * CC + n0 + c;
#pragma unroll
      for (int i = 0; i < 16; i += 4) {
        v4f f = *(const v4f*)(src + i);
#pragma unroll
        for (int q = 0; q < 4; ++q) sB[(c + i + q) * G_LD + kk] = f2bf(f[q]);
      }
    }
    __syncthreads();

    v16bf af[2];
#pragma unroll
    for (int i = 0; i < 2; ++i)
      af[i] = load_frag_a(sA + (wm * 32 + i * 16 + nn) * G_LD, half);
#pragma unroll
    for (int j = 0; j < 4; ++j) {
      v16bf bf_ = load_frag_b(sB + (wn * 64 + j * 16 + nn) * G_LD, half);
#pragma unroll
      for (int i = 0; i < 2; ++i) acc[i][j] = wmma_bf16(af[i], bf_, acc[i][j]);
    }
  }

#pragma unroll
  for (int i = 0; i < 2; ++i)
#pragma unroll
    for (int j = 0; j < 4; ++j)
#pragma unroll
      for (int r = 0; r < 8; ++r) {
        const int mg = m0 + wm * 32 + i * 16 + half * 8 + r;
        const int ng = n0 + wn * 64 + j * 16 + nn;
        out[(size_t)mg * CC + ng] = acc[i][j][r] + bias[ng];
      }
}

// ---------------------------------------------------------------------------
extern "C" void kernel_launch(void* const* d_in, const int* in_sizes, int n_in,
                              void* d_out, int out_size, void* d_ws, size_t ws_size,
                              hipStream_t stream) {
  (void)in_sizes; (void)n_in; (void)out_size; (void)ws_size;
  const float* x      = (const float*)d_in[0];
  const float* w_attn = (const float*)d_in[1];
  const float* b_attn = (const float*)d_in[2];
  const float* w_proj = (const float*)d_in[3];
  const float* b_proj = (const float*)d_in[4];
  float* out = (float*)d_out;

  const size_t BHTD = (size_t)BB * NH * TT * HD;      // 8.4M elems
  bf16_t* Q = (bf16_t*)d_ws;
  bf16_t* K = Q + BHTD;
  bf16_t* V = K + BHTD;
  bf16_t* Y = V + BHTD;                               // [B,T,C] bf16

  dim3 g1(MTOT / G_BM, C3 / G_BM);                    // 64 x 24
  qkv_gemm<<<g1, 256, 0, stream>>>(x, w_attn, b_attn, Q, K, V);

  dim3 g2(TT / 64, BB * NH);                          // 32 x 64
  flash_attn<<<g2, 128, 0, stream>>>(Q, K, V, Y);

  dim3 g3(MTOT / G_BM, CC / G_BM);                    // 64 x 8
  proj_gemm<<<g3, 256, 0, stream>>>(Y, w_proj, b_proj, out);
}
